// AttentionFFM_31997506355237
// MI455X (gfx1250) — matrix operand order repaired and checked by reading, not compiled, not run on hardware
//
#include <hip/hip_runtime.h>

// B=1024, M=64, K=16, K_RANK=16
// x : (B, 64, 16) f32   d_in[0]
// vk: (64, 16)    f32   d_in[1]
// out:(B, 64, 16) f32   d_out
// d_ws: holds w = vk @ vk^T  (64*64 f32 = 16 KB)

typedef __attribute__((ext_vector_type(16))) _Float16 v16h;
typedef __attribute__((ext_vector_type(8)))  float    v8f;

// ---------------------------------------------------------------------------
// Kernel 1: w = vk @ vk^T  (64x64) via V_WMMA_F32_16X16X32_F16.
// 16 waves, one 16x16 output tile per wave. K=16 is zero-padded to the
// instruction's K=32. Launch: <<<1, 512>>> (all lanes active: EXEC==all-1s,
// required for WMMA).
// ---------------------------------------------------------------------------
__global__ void ffm_make_w_wmma(const float* __restrict__ vk,
                                float* __restrict__ w) {
    const int tid  = threadIdx.x;
    const int lane = tid & 31;
    const int wave = tid >> 5;      // 0..15
    const int ti   = wave >> 2;     // row tile 0..3
    const int tj   = wave & 3;      // col tile 0..3
    const int hsel = lane >> 4;     // 0: lanes 0-15, 1: lanes 16-31
    const int lrow = lane & 15;

    // A: 16x32 f16, rows M = 16*ti + lrow.
    // ISA layout: VGPR0-3 hold K=0..7 (lanes 0-15) / K=8..15 (lanes 16-31);
    // VGPR4-7 hold K=16..31 -> zero padding (real K ends at 16).
    const float* arow = vk + (16 * ti + lrow) * 16;
    v16h a;
#pragma unroll
    for (int e = 0; e < 8; ++e)  a[e] = (_Float16)arow[8 * hsel + e];
#pragma unroll
    for (int e = 8; e < 16; ++e) a[e] = (_Float16)0.0f;

    // B: 32x16 f16, cols N = 16*tj + lrow.
    // Same K layout as A: elements 0-7 hold K=8*hsel+e, elements 8-15
    // hold K=16..31 -> zero padding.  B[k][n] = vk[n][k].
    const float* brow = vk + (16 * tj + lrow) * 16;
    v16h bm;
#pragma unroll
    for (int e = 0; e < 16; ++e)
        bm[e] = (e >= 8) ? (_Float16)0.0f : (_Float16)brow[8 * hsel + e];

    v8f c = {};
    c = __builtin_amdgcn_wmma_f32_16x16x32_f16(
            /*neg_a=*/false, a, /*neg_b=*/false, bm,
            /*c_mod=*/(short)0, c, /*reuse_a=*/false, /*reuse_b=*/false);

    // D layout: VGPR r, lane l -> M = r + 8*(l/16), N = l%16.
#pragma unroll
    for (int r = 0; r < 8; ++r) {
        const int M = r + 8 * hsel;
        w[(16 * ti + M) * 64 + (16 * tj + lrow)] = c[r];
    }
}

// ---------------------------------------------------------------------------
// Kernel 2: per-batch attention.  One 256-thread block (8 wave32s) per b.
// LDS: x[b] transposed to sx[k][i] (rows padded to 65 floats) + w rows
// padded to 65 floats -> conflict-free reads.  Each thread owns (i, k) with
// k = tid&15 so output stores are coalesced.
// ---------------------------------------------------------------------------
__global__ void __launch_bounds__(256)
ffm_attention(const float* __restrict__ x, const float* __restrict__ w,
              float* __restrict__ out) {
    __shared__ float sx[16 * 65];   // sx[k*65 + i] = x[b][i][k]
    __shared__ float sw[64 * 65];   // sw[i*65 + j] = w[i][j]

    const int b   = blockIdx.x;
    const int tid = threadIdx.x;
    const float* xb = x + b * 1024;

    __builtin_prefetch(xb, 0, 0);   // global_prefetch_b8: warm the x tile

#pragma unroll
    for (int idx = tid; idx < 1024; idx += 256) {
        const int i = idx >> 4, k = idx & 15;
        sx[k * 65 + i] = xb[idx];
    }
#pragma unroll
    for (int idx = tid; idx < 4096; idx += 256) {
        const int i = idx >> 6, j = idx & 63;
        sw[i * 65 + j] = w[idx];
    }
    __syncthreads();

    const int k  = tid & 15;        // fastest output dim -> coalesced stores
    const int i0 = tid >> 4;        // 0..15
    float* ob = out + b * 1024;
    const float* vrow = &sx[k * 65];

#pragma unroll
    for (int ii = 0; ii < 4; ++ii) {
        const int i = i0 + 16 * ii;
        const float vi = vrow[i];
        const float* wrow = &sw[i * 65];

        // pass 1: row max for a stable softmax
        float m = -3.402823466e38f;
#pragma unroll 8
        for (int j = 0; j < 64; ++j) {
            const float t = vi * vrow[j] * wrow[j];
            m = fmaxf(m, t);
        }
        // pass 2: sum of exp and exp-weighted sum of v[j]
        float se = 0.0f, sv = 0.0f;
#pragma unroll 8
        for (int j = 0; j < 64; ++j) {
            const float vj = vrow[j];
            const float e  = __expf(vi * vj * wrow[j] - m);
            se += e;
            sv += e * vj;
        }
        ob[i * 16 + k] = vi * sv / se;
    }
}

extern "C" void kernel_launch(void* const* d_in, const int* in_sizes, int n_in,
                              void* d_out, int out_size, void* d_ws, size_t ws_size,
                              hipStream_t stream) {
    const float* x  = (const float*)d_in[0];   // (1024, 64, 16)
    const float* vk = (const float*)d_in[1];   // (64, 16)
    float* out = (float*)d_out;                // (1024, 64, 16)
    float* w   = (float*)d_ws;                 // 64*64 floats scratch

    ffm_make_w_wmma<<<1, 512, 0, stream>>>(vk, w);
    ffm_attention<<<1024, 256, 0, stream>>>(x, w, out);
}
